// cross_attention_88416196755895
// MI455X (gfx1250) — matrix operand order repaired and checked by reading, not compiled, not run on hardware
//
#include <hip/hip_runtime.h>
#include <hip/hip_bf16.h>

typedef __bf16 bf16_t;
typedef __attribute__((ext_vector_type(16))) __bf16 v16bf;
typedef __attribute__((ext_vector_type(8)))  float  v8f;

#define C_CH  32
#define HW    252
#define LTOK  1024
#define DFEAT 2048
#define NB    8

__device__ __forceinline__ int reflect_idx(int i) { return i < HW ? i : (2 * HW - 2 - i); }

// ---------------------------------------------------------------------------
// Kernel 1: per-feature inverse L2 norms over the 1024 (padded) tokens.
// One wave per (tensor, batch, feature). Feature (c,h1,w1) == contiguous
// 32x32 block at (h1*32, w1*32) of channel c (reflect pad at the edges).
// ---------------------------------------------------------------------------
__global__ __launch_bounds__(256) void norm_kernel(const float* __restrict__ q,
                                                   const float* __restrict__ k,
                                                   const float* __restrict__ v,
                                                   float* __restrict__ inv_norms)
{
    const int gw   = blockIdx.x * 8 + (threadIdx.x >> 5);   // global wave id
    const int lane = threadIdx.x & 31;                      // h2
    const int d    = gw & (DFEAT - 1);
    const int rest = gw >> 11;
    const int b    = rest & 7;
    const int t    = rest >> 3;                             // 0=q 1=k 2=v
    const float* src = (t == 0) ? q : (t == 1) ? k : v;

    const int c  = d >> 6;
    const int h1 = (d >> 3) & 7;
    const int w1 = d & 7;
    const int sh = reflect_idx(h1 * 32 + lane);
    const float* rowp = src + (((size_t)b * C_CH + c) * HW + sh) * HW;

    float s = 0.0f;
    #pragma unroll 4
    for (int w2 = 0; w2 < 32; ++w2) {
        const int sw = reflect_idx(w1 * 32 + w2);
        const float x = rowp[sw];
        s += x * x;
    }
    #pragma unroll
    for (int o = 16; o > 0; o >>= 1) s += __shfl_xor(s, o, 32);
    if (lane == 0) inv_norms[gw] = 1.0f / fmaxf(sqrtf(s), 1e-12f);
}

// ---------------------------------------------------------------------------
// Kernel 2: gather + normalize + convert to bf16.
// transposed==0 : dst[b][l][d]   (Qn, Kn — row-major over d)
// transposed==1 : dst[b][d][l]   (Vnt — row-major over m, for GEMM2 B-frags)
// ---------------------------------------------------------------------------
__global__ __launch_bounds__(256) void gather_kernel(const float* __restrict__ src,
                                                     const float* __restrict__ invn,
                                                     bf16_t* __restrict__ dst,
                                                     int transposed)
{
    const int b = blockIdx.z;
    const size_t idx = (size_t)blockIdx.x * 256 + threadIdx.x;   // < 2048*1024
    int l, d;
    if (!transposed) { d = (int)(idx & (DFEAT - 1)); l = (int)(idx >> 11); }
    else             { l = (int)(idx & (LTOK  - 1)); d = (int)(idx >> 10); }

    const int h2 = l >> 5, w2 = l & 31;
    const int c  = d >> 6, h1 = (d >> 3) & 7, w1 = d & 7;
    const int sh = reflect_idx(h1 * 32 + h2);
    const int sw = reflect_idx(w1 * 32 + w2);
    const float x = src[(((size_t)b * C_CH + c) * HW + sh) * HW + sw];
    dst[(size_t)b * ((size_t)LTOK * DFEAT) + idx] = (bf16_t)(x * invn[((size_t)b << 11) | d]);
}

// ---------------------------------------------------------------------------
// WMMA fragment loaders (bf16 16x16x32, wave32 layouts per CDNA5 ISA 7.12.2).
// A (16x32): lane r=lane%16 holds row M=r; half h=lane/16:
//   vector elems 0-7  = K = 8h..8h+7, elems 8-15 = K = 16+8h..23+8h
// B (32x16): lane n=lane%16 holds col N=n; half h:
//   vector elems 0-7  = K = 8h..8h+7, elems 8-15 = K = 16+8h..23+8h (same as A)
// Both are two 16B LDS loads per lane from row-major [rows][32] slabs.
// ---------------------------------------------------------------------------
__device__ __forceinline__ v16bf load_frag_a(const bf16_t* base, int r, int h)
{
    union { v16bf v; uint4 u[2]; } t;
    const uint4* p = reinterpret_cast<const uint4*>(base + r * 32);
    t.u[0] = p[h];
    t.u[1] = p[2 + h];
    return t.v;
}
__device__ __forceinline__ v16bf load_frag_b(const bf16_t* base, int n, int h)
{
    union { v16bf v; uint4 u[2]; } t;
    const uint4* p = reinterpret_cast<const uint4*>(base + n * 32);
    t.u[0] = p[h];
    t.u[1] = p[2 + h];
    return t.v;
}

// ---------------------------------------------------------------------------
// NT-GEMM core, 128x128 workgroup tile, double-buffered LDS, one barrier/step.
// 8 waves in a 4x2 grid; each wave computes 32x64 (2x4 WMMA tiles, 8 acc).
// Per k-step(32): issue next global slab, 6 fragment loads + 8 WMMAs from the
// current buffer, store next slab into the other buffer, single barrier.
// ---------------------------------------------------------------------------
__device__ __forceinline__ void gemm_core(const bf16_t* __restrict__ Ab,
                                          const bf16_t* __restrict__ Bb,
                                          int lda, int ldb, int Kdim,
                                          int m0, int n0,
                                          bf16_t (&As)[2][128][32],
                                          bf16_t (&Bs)[2][128][32],
                                          v8f (&acc)[2][4])
{
    const int tid  = threadIdx.x;
    const int lane = tid & 31, wave = tid >> 5;
    const int wr = wave >> 1, wc = wave & 1;       // 4x2 wave grid
    const int r = lane & 15,  h  = lane >> 4;
    const int srow = tid >> 2;                     // 0..63
    const int scol = (tid & 3) * 8;                // 16B chunk within 64B row

    const bf16_t* aP0 = Ab + (size_t)(m0 + srow)      * lda + scol;
    const bf16_t* aP1 = Ab + (size_t)(m0 + srow + 64) * lda + scol;
    const bf16_t* bP0 = Bb + (size_t)(n0 + srow)      * ldb + scol;
    const bf16_t* bP1 = Bb + (size_t)(n0 + srow + 64) * ldb + scol;

    // prologue: fill buffer 0 with k=0 slab
    {
        const uint4 a0 = *reinterpret_cast<const uint4*>(aP0);
        const uint4 a1 = *reinterpret_cast<const uint4*>(aP1);
        const uint4 b0 = *reinterpret_cast<const uint4*>(bP0);
        const uint4 b1 = *reinterpret_cast<const uint4*>(bP1);
        *reinterpret_cast<uint4*>(&As[0][srow][scol])      = a0;
        *reinterpret_cast<uint4*>(&As[0][srow + 64][scol]) = a1;
        *reinterpret_cast<uint4*>(&Bs[0][srow][scol])      = b0;
        *reinterpret_cast<uint4*>(&Bs[0][srow + 64][scol]) = b1;
    }
    __syncthreads();

    int buf = 0;
    for (int k = 0; k < Kdim; k += 32, buf ^= 1) {
        const int kn = k + 32;
        const bool more = kn < Kdim;
        uint4 na0, na1, nb0, nb1;
        if (more) {                                  // global loads for k+1 slab
            na0 = *reinterpret_cast<const uint4*>(aP0 + kn);
            na1 = *reinterpret_cast<const uint4*>(aP1 + kn);
            nb0 = *reinterpret_cast<const uint4*>(bP0 + kn);
            nb1 = *reinterpret_cast<const uint4*>(bP1 + kn);
            if (kn + 32 < Kdim) {                    // keep caches one step ahead
                __builtin_prefetch(aP0 + kn + 32, 0, 2);
                __builtin_prefetch(bP0 + kn + 32, 0, 2);
            }
        }

        // compute on current buffer: 2 A-frags x 4 B-frags -> 8 WMMAs
        v16bf fa[2], fb[4];
        fa[0] = load_frag_a(&As[buf][wr * 32][0],      r, h);
        fa[1] = load_frag_a(&As[buf][wr * 32 + 16][0], r, h);
        #pragma unroll
        for (int j = 0; j < 4; ++j)
            fb[j] = load_frag_b(&Bs[buf][wc * 64 + j * 16][0], r, h);
        #pragma unroll
        for (int i = 0; i < 2; ++i)
            #pragma unroll
            for (int j = 0; j < 4; ++j)
                acc[i][j] = __builtin_amdgcn_wmma_f32_16x16x32_bf16(
                    false, fa[i], false, fb[j], (short)0, acc[i][j], false, false);

        if (more) {                                  // fill the other buffer
            *reinterpret_cast<uint4*>(&As[buf ^ 1][srow][scol])      = na0;
            *reinterpret_cast<uint4*>(&As[buf ^ 1][srow + 64][scol]) = na1;
            *reinterpret_cast<uint4*>(&Bs[buf ^ 1][srow][scol])      = nb0;
            *reinterpret_cast<uint4*>(&Bs[buf ^ 1][srow + 64][scol]) = nb1;
        }
        __syncthreads();
    }
}

// ---------------------------------------------------------------------------
// Kernel 3: S = (Qn . Kn^T) * 2048^-0.5   (per batch, 1024x1024 f32)
// ---------------------------------------------------------------------------
__global__ __launch_bounds__(256) void gemm1_kernel(const bf16_t* __restrict__ Qn,
                                                    const bf16_t* __restrict__ Kn,
                                                    float* __restrict__ S, float scale)
{
    __shared__ bf16_t As[2][128][32];
    __shared__ bf16_t Bs[2][128][32];
    const int b = blockIdx.z;
    const bf16_t* Ab = Qn + (size_t)b * LTOK * DFEAT;
    const bf16_t* Bb = Kn + (size_t)b * LTOK * DFEAT;
    float* Cb = S + (size_t)b * LTOK * LTOK;
    const int m0 = blockIdx.y * 128, n0 = blockIdx.x * 128;

    v8f acc[2][4];
    #pragma unroll
    for (int i = 0; i < 2; ++i)
        #pragma unroll
        for (int j = 0; j < 4; ++j)
            acc[i][j] = (v8f){0, 0, 0, 0, 0, 0, 0, 0};

    gemm_core(Ab, Bb, DFEAT, DFEAT, DFEAT, m0, n0, As, Bs, acc);

    const int lane = threadIdx.x & 31, wave = threadIdx.x >> 5;
    const int wr = wave >> 1, wc = wave & 1;
    const int r = lane & 15,  h  = lane >> 4;
    #pragma unroll
    for (int i = 0; i < 2; ++i) {
        const int row = m0 + wr * 32 + i * 16 + h * 8;   // C: VGPR v -> M = v(+8h)
        #pragma unroll
        for (int j = 0; j < 4; ++j) {
            const int col = n0 + wc * 64 + j * 16 + r;
            #pragma unroll
            for (int v = 0; v < 8; ++v)
                Cb[(size_t)(row + v) * LTOK + col] = acc[i][j][v] * scale;
        }
    }
}

// ---------------------------------------------------------------------------
// Kernel 4: row softmax, f32 S -> bf16 P (one block per row of 1024)
// ---------------------------------------------------------------------------
__global__ __launch_bounds__(256) void softmax_kernel(const float* __restrict__ S,
                                                      bf16_t* __restrict__ P)
{
    __shared__ float red[256];
    const int b = blockIdx.z, row = blockIdx.x;
    const float* s = S + ((size_t)b * LTOK + row) * LTOK;
    bf16_t*      p = P + ((size_t)b * LTOK + row) * LTOK;
    const int tid = threadIdx.x;

    const float4 x = reinterpret_cast<const float4*>(s)[tid];
    float m = fmaxf(fmaxf(x.x, x.y), fmaxf(x.z, x.w));
    red[tid] = m; __syncthreads();
    for (int o = 128; o > 0; o >>= 1) { if (tid < o) red[tid] = fmaxf(red[tid], red[tid + o]); __syncthreads(); }
    m = red[0]; __syncthreads();

    const float e0 = __expf(x.x - m), e1 = __expf(x.y - m);
    const float e2 = __expf(x.z - m), e3 = __expf(x.w - m);
    red[tid] = e0 + e1 + e2 + e3; __syncthreads();
    for (int o = 128; o > 0; o >>= 1) { if (tid < o) red[tid] += red[tid + o]; __syncthreads(); }
    const float inv = 1.0f / red[0];

    p[tid * 4 + 0] = (bf16_t)(e0 * inv);
    p[tid * 4 + 1] = (bf16_t)(e1 * inv);
    p[tid * 4 + 2] = (bf16_t)(e2 * inv);
    p[tid * 4 + 3] = (bf16_t)(e3 * inv);
}

// ---------------------------------------------------------------------------
// Kernel 5: O = P . Vn  via Vnt[d][m]; epilogue scatters straight into the
// inverse-permuted, cropped (b,c,252,252) output (pad rows/cols dropped).
// ---------------------------------------------------------------------------
__global__ __launch_bounds__(256) void gemm2_kernel(const bf16_t* __restrict__ P,
                                                    const bf16_t* __restrict__ Vnt,
                                                    float* __restrict__ out)
{
    __shared__ bf16_t As[2][128][32];
    __shared__ bf16_t Bs[2][128][32];
    const int b = blockIdx.z;
    const bf16_t* Ab = P   + (size_t)b * LTOK * LTOK;
    const bf16_t* Bb = Vnt + (size_t)b * DFEAT * LTOK;
    const int m0 = blockIdx.y * 128, n0 = blockIdx.x * 128;

    v8f acc[2][4];
    #pragma unroll
    for (int i = 0; i < 2; ++i)
        #pragma unroll
        for (int j = 0; j < 4; ++j)
            acc[i][j] = (v8f){0, 0, 0, 0, 0, 0, 0, 0};

    gemm_core(Ab, Bb, LTOK, LTOK, LTOK, m0, n0, As, Bs, acc);

    const int lane = threadIdx.x & 31, wave = threadIdx.x >> 5;
    const int wr = wave >> 1, wc = wave & 1;
    const int r = lane & 15,  h  = lane >> 4;
    #pragma unroll
    for (int i = 0; i < 2; ++i) {
        const int row = m0 + wr * 32 + i * 16 + h * 8;
        #pragma unroll
        for (int j = 0; j < 4; ++j) {
            const int d  = n0 + wc * 64 + j * 16 + r;
            const int c  = d >> 6, h1 = (d >> 3) & 7, w1 = d & 7;
            #pragma unroll
            for (int v = 0; v < 8; ++v) {
                const int l  = row + v;
                const int h2 = l >> 5, w2 = l & 31;
                const int hh = h1 * 32 + h2, ww = w1 * 32 + w2;
                if (hh < HW && ww < HW)
                    out[(((size_t)b * C_CH + c) * HW + hh) * HW + ww] = acc[i][j][v];
            }
        }
    }
}

// ---------------------------------------------------------------------------
extern "C" void kernel_launch(void* const* d_in, const int* in_sizes, int n_in,
                              void* d_out, int out_size, void* d_ws, size_t ws_size,
                              hipStream_t stream)
{
    const float* q = (const float*)d_in[0];
    const float* k = (const float*)d_in[1];
    const float* v = (const float*)d_in[2];
    float* out = (float*)d_out;
    char*  ws  = (char*)d_ws;

    const size_t offQ = 0;
    const size_t offK = offQ + (size_t)NB * LTOK * DFEAT * 2;   // Qn bf16
    const size_t offV = offK + (size_t)NB * LTOK * DFEAT * 2;   // Kn bf16
    const size_t offP = offV + (size_t)NB * DFEAT * LTOK * 2;   // Vnt bf16
    const size_t offS = offP + (size_t)NB * LTOK * LTOK * 2;    // P bf16
    const size_t offN = offS + (size_t)NB * LTOK * LTOK * 4;    // S f32

    bf16_t* Qn    = (bf16_t*)(ws + offQ);
    bf16_t* Kn    = (bf16_t*)(ws + offK);
    bf16_t* Vnt   = (bf16_t*)(ws + offV);
    bf16_t* Pm    = (bf16_t*)(ws + offP);
    float*  S     = (float*)(ws + offS);
    float*  norms = (float*)(ws + offN);

    norm_kernel<<<dim3(6144), 256, 0, stream>>>(q, k, v, norms);

    gather_kernel<<<dim3(8192, 1, NB), 256, 0, stream>>>(q, norms + (size_t)0 * NB * DFEAT, Qn, 0);
    gather_kernel<<<dim3(8192, 1, NB), 256, 0, stream>>>(k, norms + (size_t)1 * NB * DFEAT, Kn, 0);
    gather_kernel<<<dim3(8192, 1, NB), 256, 0, stream>>>(v, norms + (size_t)2 * NB * DFEAT, Vnt, 1);

    const float scale = 0.022097086912079608f;   // 2048^-0.5
    gemm1_kernel<<<dim3(8, 8, NB), 256, 0, stream>>>(Qn, Kn, S, scale);
    softmax_kernel<<<dim3(1024, 1, NB), 256, 0, stream>>>(S, Pm);
    gemm2_kernel<<<dim3(16, 8, NB), 256, 0, stream>>>(Pm, Vnt, out);

    (void)in_sizes; (void)n_in; (void)out_size; (void)ws_size;
}
